// MultiHeadAttention_32143535243926
// MI455X (gfx1250) — compile-verified, matrix operand order checked by reading, not run on hardware
//
#include <hip/hip_runtime.h>
#include <hip/hip_bf16.h>

// ---------------------------------------------------------------------------
// MHA forward for B=2, S=2048, D=1024, H=16, DH=64 on gfx1250 (wave32, WMMA)
// ---------------------------------------------------------------------------

typedef __attribute__((ext_vector_type(16))) _Float16 v16h;
typedef __attribute__((ext_vector_type(8)))  _Float16 v8h;
typedef __attribute__((ext_vector_type(8)))  float    v8f;

union frag16 { v16h f; v8h h[2]; };

template <bool V> struct BoolC { static constexpr bool value = V; };

__device__ __forceinline__ v8f wmma_f16(v16h a, v16h b, v8f c) {
    // D = A(16x32 f16) * B(32x16 f16) + C(16x16 f32)
    return __builtin_amdgcn_wmma_f32_16x16x32_f16(
        /*neg_a=*/false, a, /*neg_b=*/false, b,
        /*c_mod=*/(short)0, c, /*reuse_a=*/false, /*reuse_b=*/false);
}

// ---- 16-lane row reductions -----------------------------------------------
// xor-by-{1,2,4,8} butterfly stays inside a DPP16 row: quad_perm for 1/2,
// row_half_mirror (0x141) for 4, row_mirror (0x140) for 8. Full VALU rate,
// no LDS pipe traffic.
#if __has_builtin(__builtin_amdgcn_mov_dpp)
template <int CTRL>
__device__ __forceinline__ float dpp_f32(float x) {
    int i = __builtin_bit_cast(int, x);
    i = __builtin_amdgcn_mov_dpp(i, CTRL, 0xf, 0xf, true);
    return __builtin_bit_cast(float, i);
}
__device__ __forceinline__ float red_max16(float x) {
    x = fmaxf(x, dpp_f32<0xB1>(x));   // quad_perm(1,0,3,2)  : xor 1
    x = fmaxf(x, dpp_f32<0x4E>(x));   // quad_perm(2,3,0,1)  : xor 2
    x = fmaxf(x, dpp_f32<0x141>(x));  // row_half_mirror     : xor 4
    x = fmaxf(x, dpp_f32<0x140>(x));  // row_mirror          : xor 8
    return x;
}
__device__ __forceinline__ float red_sum16(float x) {
    x += dpp_f32<0xB1>(x);
    x += dpp_f32<0x4E>(x);
    x += dpp_f32<0x141>(x);
    x += dpp_f32<0x140>(x);
    return x;
}
#else
__device__ __forceinline__ float red_max16(float x) {
    for (int d = 1; d < 16; d <<= 1) x = fmaxf(x, __shfl_xor(x, d, 32));
    return x;
}
__device__ __forceinline__ float red_sum16(float x) {
    for (int d = 1; d < 16; d <<= 1) x += __shfl_xor(x, d, 32);
    return x;
}
#endif

// Load a 16x32 f16 operand tile (A or B pattern, identical per ISA 7.12.2)
// from a row-major f16 matrix: lane L holds row (row0 + (L&15)), two 16B
// chunks at k-offsets g*8 and 16+g*8 with g = L>>4.
__device__ __forceinline__ v16h load_frag_h(const _Float16* __restrict__ base,
                                            int ld, int row0, int k0) {
    const int lane = threadIdx.x & 31;
    const int r = lane & 15, g = lane >> 4;
    const _Float16* p = base + (size_t)(row0 + r) * ld + k0 + g * 8;
    frag16 u;
    u.h[0] = *(const v8h*)(p);
    u.h[1] = *(const v8h*)(p + 16);
    return u.f;
}

// Same pattern but source is f32: load and convert in-register.
__device__ __forceinline__ v16h load_frag_f32(const float* __restrict__ base,
                                              int ld, int row0, int k0) {
    const int lane = threadIdx.x & 31;
    const int r = lane & 15, g = lane >> 4;
    const float* p = base + (size_t)(row0 + r) * ld + k0 + g * 8;
    frag16 u;
#pragma unroll
    for (int i = 0; i < 8; ++i) u.h[0][i] = (_Float16)p[i];
#pragma unroll
    for (int i = 0; i < 8; ++i) u.h[1][i] = (_Float16)p[16 + i];
    return u.f;
}

// ---------------------------------------------------------------------------
// Stage 0: f32 -> f16 weight conversion
// ---------------------------------------------------------------------------
__global__ void cvt_f32_f16(const float* __restrict__ src,
                            _Float16* __restrict__ dst, int n) {
    int i = blockIdx.x * blockDim.x + threadIdx.x;
    if (i < n) dst[i] = (_Float16)src[i];
}

// ---------------------------------------------------------------------------
// GEMM: Y[M=4096, N=1024] = X[4096x1024] * W^T  (W is [1024(N) x 1024(K)])
// IN_F32:    1 -> X is f32 (converted in-register), 0 -> X is f16
// STORE_MODE 0 -> f16 row-major (ld 1024)
//            1 -> f16 per-head transposed: Vt[((b*16+h)*64+dh)*2048 + s]
//            2 -> f32 row-major (final output)
// Block: 256 threads = 8 waves (2 M x 4 N), block tile 64x256,
// wave tile 32x64: two A frags share four B frags -> 8 WMMA / 12 loads.
// ---------------------------------------------------------------------------
template <int IN_F32, int STORE_MODE>
__global__ __launch_bounds__(256)
void gemm_xwT(const void* __restrict__ Xv, const _Float16* __restrict__ W,
              void* __restrict__ Yv) {
    const int lane = threadIdx.x & 31;
    const int wave = threadIdx.x >> 5;
    const int wm = wave >> 2, wn = wave & 3;
    const int row0 = (blockIdx.x & 63) * 64 + wm * 32;   // 64 M-blocks
    const int col0 = (blockIdx.x >> 6) * 256 + wn * 64;  // 4 N-blocks

    v8f acc[2][4] = {};

    for (int k = 0; k < 1024; k += 32) {
        v16h a0, a1;
        if constexpr (IN_F32) {
            a0 = load_frag_f32((const float*)Xv, 1024, row0, k);
            a1 = load_frag_f32((const float*)Xv, 1024, row0 + 16, k);
        } else {
            a0 = load_frag_h((const _Float16*)Xv, 1024, row0, k);
            a1 = load_frag_h((const _Float16*)Xv, 1024, row0 + 16, k);
        }
#pragma unroll
        for (int t = 0; t < 4; ++t) {
            v16h b = load_frag_h(W, 1024, col0 + t * 16, k);
            acc[0][t] = wmma_f16(a0, b, acc[0][t]);
            acc[1][t] = wmma_f16(a1, b, acc[1][t]);
        }
    }

    // C layout: lane L -> col = (L&15), row = (L>>4)*8 + j  (j = VGPR index)
    const int r = lane & 15, g = lane >> 4;
#pragma unroll
    for (int m = 0; m < 2; ++m) {
#pragma unroll
        for (int t = 0; t < 4; ++t) {
#pragma unroll
            for (int j = 0; j < 8; ++j) {
                const int row = row0 + m * 16 + g * 8 + j;
                const int col = col0 + t * 16 + r;
                if constexpr (STORE_MODE == 0) {
                    ((_Float16*)Yv)[(size_t)row * 1024 + col] =
                        (_Float16)acc[m][t][j];
                } else if constexpr (STORE_MODE == 1) {
                    const int bb = row >> 11, s = row & 2047;
                    const int h = col >> 6, dh = col & 63;
                    ((_Float16*)Yv)[((size_t)((bb * 16 + h) * 64 + dh)) * 2048 + s] =
                        (_Float16)acc[m][t][j];
                } else {
                    ((float*)Yv)[(size_t)row * 1024 + col] = acc[m][t][j];
                }
            }
        }
    }
}

// ---------------------------------------------------------------------------
// Flash attention: one wave handles a 32-query tile (two 16-row WMMA tiles)
// for one (b, h); K/Vt fragments are shared between the two q-tiles.
// Causal masking applied ONLY on the diagonal key block; all earlier blocks
// take the fast path (no per-element compares). Next key block's K/Vt lines
// are prefetched (global_prefetch_b8) while the current block computes.
// ---------------------------------------------------------------------------
__global__ __launch_bounds__(32)
void attn_kernel(const _Float16* __restrict__ Q16,
                 const _Float16* __restrict__ K16,
                 const _Float16* __restrict__ Vt16,
                 const int* __restrict__ mask,
                 _Float16* __restrict__ O16) {
    constexpr int S = 2048, D = 1024, H = 16;
    constexpr float NEG = -1.0e9f;
    const int lane = threadIdx.x & 31;
    const int r = lane & 15, g = lane >> 4;
    const int q0 = blockIdx.x * 32;
    const int h  = blockIdx.y;
    const int b  = blockIdx.z;

    const _Float16* Qh  = Q16 + (size_t)b * S * D + h * 64;
    const _Float16* Kh  = K16 + (size_t)b * S * D + h * 64;
    const _Float16* Vth = Vt16 + ((size_t)(b * H + h) * 64) * S;
    const int* mb = mask + b * S;

    __shared__ __align__(16) _Float16 Plds[32 * 32];

    // Preload Q operand tiles: 2 q-subtiles x 2 k-steps over dh=64
    v16h aq[2][2];
#pragma unroll
    for (int qt = 0; qt < 2; ++qt) {
        aq[qt][0] = load_frag_h(Qh, D, q0 + qt * 16, 0);
        aq[qt][1] = load_frag_h(Qh, D, q0 + qt * 16, 32);
    }

    v8f acc[2][4] = {};
    float m_run[2][8], l_run[2][8];
#pragma unroll
    for (int qt = 0; qt < 2; ++qt)
#pragma unroll
        for (int j = 0; j < 8; ++j) { m_run[qt][j] = -3.0e38f; l_run[qt][j] = 0.0f; }

    const float scale = 0.125f;  // 1/sqrt(DH=64)

    auto process_block = [&](const int key0, auto causal_c) {
        constexpr bool CAUSAL = decltype(causal_c)::value;

        // K fragments for this 32-key block (shared across both q-subtiles)
        v16h bK[2][2];
#pragma unroll
        for (int kt = 0; kt < 2; ++kt) {
            bK[kt][0] = load_frag_h(Kh, D, key0 + kt * 16, 0);
            bK[kt][1] = load_frag_h(Kh, D, key0 + kt * 16, 32);
        }

        // Prefetch next key block while this one computes: 32 lanes cover the
        // 32 next K rows; 2 more prefetches cover all 64 Vt rows at the next
        // key offset. Speculative past the causal frontier is fine (dropped
        // on fault; lines are needed by neighboring q-tiles anyway).
        {
            const int nk = key0 + 32;
            if (nk < S) {
                __builtin_prefetch(Kh + (size_t)(nk + lane) * D, 0, 1);
                __builtin_prefetch(Vth + (size_t)lane * S + nk, 0, 1);
                __builtin_prefetch(Vth + (size_t)(lane + 32) * S + nk, 0, 1);
            }
        }

        const int key_a = key0 + r;
        const int key_b = key0 + 16 + r;
        const float pad_a = (mb[key_a] == 0) ? NEG : 0.0f;
        const float pad_b = (mb[key_b] == 0) ? NEG : 0.0f;

#pragma unroll
        for (int qt = 0; qt < 2; ++qt) {
            // scores: 16 q x 32 keys, reduce over dh=64
            v8f c0 = {}, c1 = {};
            c0 = wmma_f16(aq[qt][0], bK[0][0], c0);
            c0 = wmma_f16(aq[qt][1], bK[0][1], c0);
            c1 = wmma_f16(aq[qt][0], bK[1][0], c1);
            c1 = wmma_f16(aq[qt][1], bK[1][1], c1);

            // softmax; row j lives across the 16 lanes of group g
#pragma unroll
            for (int j = 0; j < 8; ++j) {
                float s0 = c0[j] * scale + pad_a;
                float s1 = c1[j] * scale + pad_b;
                if constexpr (CAUSAL) {
                    const int q = q0 + qt * 16 + g * 8 + j;
                    if (key_a > q) s0 = NEG + pad_a;
                    if (key_b > q) s1 = NEG + pad_b;
                }
                const float rm    = red_max16(fmaxf(s0, s1));
                const float m_new = fmaxf(m_run[qt][j], rm);
                const float resc  = __expf(m_run[qt][j] - m_new);
                const float p0 = __expf(s0 - m_new);
                const float p1 = __expf(s1 - m_new);
                const float rs = red_sum16(p0 + p1);
                l_run[qt][j] = l_run[qt][j] * resc + rs;
                m_run[qt][j] = m_new;
#pragma unroll
                for (int t = 0; t < 4; ++t) acc[qt][t][j] *= resc;
                // P to LDS, row-major [32 q][32 keys]
                Plds[(qt * 16 + g * 8 + j) * 32 + r]      = (_Float16)p0;
                Plds[(qt * 16 + g * 8 + j) * 32 + 16 + r] = (_Float16)p1;
            }
        }

        // LDS stores -> cross-lane reload; DS pipe is in-order per wave but
        // wait explicitly before re-reading.
        asm volatile("s_wait_dscnt 0" ::: "memory");

        // V fragments for this key block (shared across both q-subtiles)
        v16h bV[4];
#pragma unroll
        for (int t = 0; t < 4; ++t)
            bV[t] = load_frag_h(Vth, S, t * 16, key0);

#pragma unroll
        for (int qt = 0; qt < 2; ++qt) {
            // P as A-operand (lane = query row, keys striped)
            frag16 up;
            up.h[0] = *(const v8h*)(&Plds[(qt * 16 + r) * 32 + g * 8]);
            up.h[1] = *(const v8h*)(&Plds[(qt * 16 + r) * 32 + 16 + g * 8]);
            const v16h ap = up.f;
#pragma unroll
            for (int t = 0; t < 4; ++t)
                acc[qt][t] = wmma_f16(ap, bV[t], acc[qt][t]);
        }
    };

    // full (never-masked) key blocks, then the single diagonal block
    const int kb_diag = q0 >> 5;
    for (int kb = 0; kb < kb_diag; ++kb) process_block(kb * 32, BoolC<false>{});
    process_block(kb_diag * 32, BoolC<true>{});

    // epilogue: divide by softmax denominator, write f16 [B,S,D]
#pragma unroll
    for (int qt = 0; qt < 2; ++qt) {
#pragma unroll
        for (int j = 0; j < 8; ++j) {
            const float inv = 1.0f / l_run[qt][j];
            const size_t row = (size_t)(b * S + q0 + qt * 16 + g * 8 + j);
#pragma unroll
            for (int t = 0; t < 4; ++t)
                O16[row * D + h * 64 + t * 16 + r] =
                    (_Float16)(acc[qt][t][j] * inv);
        }
    }
}

// ---------------------------------------------------------------------------
// Host-side launch
// ---------------------------------------------------------------------------
extern "C" void kernel_launch(void* const* d_in, const int* in_sizes, int n_in,
                              void* d_out, int out_size, void* d_ws, size_t ws_size,
                              hipStream_t stream) {
    const float* q    = (const float*)d_in[0];
    const float* k    = (const float*)d_in[1];
    const float* v    = (const float*)d_in[2];
    const int*   mask = (const int*)d_in[3];
    const float* Wq   = (const float*)d_in[4];
    const float* Wk   = (const float*)d_in[5];
    const float* Wv   = (const float*)d_in[6];
    const float* Wo   = (const float*)d_in[7];
    float* out = (float*)d_out;

    // workspace carve-up (f16 elements); total 20M halves = 40 MB
    _Float16* w16  = (_Float16*)d_ws;
    _Float16* Wq16 = w16;
    _Float16* Wk16 = w16 + (size_t)1  * (1 << 20);
    _Float16* Wv16 = w16 + (size_t)2  * (1 << 20);
    _Float16* Wo16 = w16 + (size_t)3  * (1 << 20);
    _Float16* Q16  = w16 + (size_t)4  * (1 << 20);
    _Float16* K16  = w16 + (size_t)8  * (1 << 20);
    _Float16* Vt16 = w16 + (size_t)12 * (1 << 20);
    _Float16* O16  = w16 + (size_t)16 * (1 << 20);

    const int NW = 1 << 20;  // 1024*1024 weight elements
    cvt_f32_f16<<<NW / 256, 256, 0, stream>>>(Wq, Wq16, NW);
    cvt_f32_f16<<<NW / 256, 256, 0, stream>>>(Wk, Wk16, NW);
    cvt_f32_f16<<<NW / 256, 256, 0, stream>>>(Wv, Wv16, NW);
    cvt_f32_f16<<<NW / 256, 256, 0, stream>>>(Wo, Wo16, NW);

    // projections: 256 blocks (64 M-tiles x 4 N-tiles), 256 threads
    gemm_xwT<1, 0><<<256, 256, 0, stream>>>(q, Wq16, Q16);
    gemm_xwT<1, 0><<<256, 256, 0, stream>>>(k, Wk16, K16);
    gemm_xwT<1, 1><<<256, 256, 0, stream>>>(v, Wv16, Vt16);  // store V^T per head

    // attention: grid = (S/32 q-tiles, H, B), one wave each
    attn_kernel<<<dim3(64, 16, 2), 32, 0, stream>>>(Q16, K16, Vt16, mask, O16);

    // output projection -> f32 d_out
    gemm_xwT<0, 2><<<256, 256, 0, stream>>>(O16, Wo16, out);
}
